// DamageConsistencyLoss_17849884082338
// MI455X (gfx1250) — matrix operand order repaired and checked by reading, not compiled, not run on hardware
//
#include <hip/hip_runtime.h>
#include <hip/hip_bf16.h>
#include <math.h>

typedef __attribute__((ext_vector_type(16))) _Float16 v16h;
typedef __attribute__((ext_vector_type(8)))  _Float16 v8h;
typedef __attribute__((ext_vector_type(8)))  float    v8f;

#define DIM_H 1024
#define DIM_W 1024
#define DIM_C 6
#define DIM_B 8
#define CH_STRIDE (DIM_H * DIM_W)

// 5x5 zero-padded box filter of a 16x16 tile (with 20x20 halo staged in a
// 32x32 zero-padded, column-major f16 LDS buffer) via two banded WMMAs:
//   vertical:   Y = A_band(16x32) * X(32x16)   (two 16-col chunks)
//   horizontal: S = Y(16x32)      * B_band(32x16)
// Values are small integers -> exact in f16, accumulated exactly in f32.
__device__ __forceinline__ v8f box_filter_16x16(const _Float16* Xc,
                                                _Float16* Ys,
                                                v16h aband, v16h bband,
                                                int lane) {
  const int n     = lane & 15;
  const int rbase = (lane < 16) ? 0 : 8;    // B-operand K rows for this lane half

  // B-operand loads: column n (and n+16), rows rbase..rbase+7 and rbase+16..rbase+23
  const v8h* c0 = (const v8h*)(Xc + n * 32 + rbase);
  const v8h* c1 = (const v8h*)(Xc + (n + 16) * 32 + rbase);
  v8h b0lo = c0[0], b0hi = c0[2];
  v8h b1lo = c1[0], b1hi = c1[2];
  v16h bop0 = __builtin_shufflevector(b0lo, b0hi, 0,1,2,3,4,5,6,7,8,9,10,11,12,13,14,15);
  v16h bop1 = __builtin_shufflevector(b1lo, b1hi, 0,1,2,3,4,5,6,7,8,9,10,11,12,13,14,15);

  v8f zc = {};
  v8f y0 = __builtin_amdgcn_wmma_f32_16x16x32_f16(false, aband, false, bop0,
                                                  (short)0, zc, false, false);
  v8f y1 = __builtin_amdgcn_wmma_f32_16x16x32_f16(false, aband, false, bop1,
                                                  (short)0, zc, false, false);

  // Stage Y (16 rows x 32 cols, row-major f16). C/D layout: VGPR r holds
  // M = r (+8 for upper lane half), N = lane%16.
  const int mb = (lane < 16) ? 0 : 8;
#pragma unroll
  for (int r = 0; r < 8; ++r) {
    Ys[(r + mb) * 32 + n]      = (_Float16)y0[r];
    Ys[(r + mb) * 32 + n + 16] = (_Float16)y1[r];
  }
  asm volatile("s_wait_dscnt 0" ::: "memory");

  // Reload Y as A-operand: lane holds row m = lane%16; K runs are
  // {0-7,16-23} for lanes 0-15 and {8-15,24-31} for lanes 16-31.
  const int k1 = (lane < 16) ? 0 : 8;
  v8h alo = *(const v8h*)(Ys + n * 32 + k1);
  v8h ahi = *(const v8h*)(Ys + n * 32 + k1 + 16);
  v16h aop = __builtin_shufflevector(alo, ahi, 0,1,2,3,4,5,6,7,8,9,10,11,12,13,14,15);

  return __builtin_amdgcn_wmma_f32_16x16x32_f16(false, aop, false, bband,
                                                (short)0, zc, false, false);
}

__global__ __launch_bounds__(256) void dcl_kernel(const float* __restrict__ pred,
                                                  const long long* __restrict__ tgt,
                                                  float* __restrict__ out) {
  // Per-wave LDS slice: X1 (32x32 f16, col-major), X2 (same), Ystage (16x32 f16)
  __shared__ __align__(16) _Float16 smem[8][2560];
  const int lane = threadIdx.x & 31;
  const int wv   = threadIdx.x >> 5;
  _Float16* X1 = smem[wv];
  _Float16* X2 = X1 + 1024;
  _Float16* Ys = X1 + 2048;

  const int tile = blockIdx.x * 8 + wv;      // 32768 tiles = 8 * 64 * 64
  const int b  = tile >> 12;
  const int ty = (tile >> 6) & 63;
  const int tx = tile & 63;
  const int y0 = ty * 16, x0 = tx * 16;

  // Constant banded matrices in A / B WMMA register layouts (per ISA 7.12.2).
  v16h aband, bband;
  {
    const int m = lane & 15;
#pragma unroll
    for (int idx = 0; idx < 16; ++idx) {
      const int j = idx >> 1, h = idx & 1;
      int base = (j < 4) ? (2 * j) : (8 + 2 * j);   // K blocks 0-7 / 16-23
      if (lane >= 16) base += 8;                    // upper half: 8-15 / 24-31
      const int K = base + h;
      aband[idx] = (K >= m && K <= m + 4) ? (_Float16)1.0f : (_Float16)0.0f;
      const int Kb = ((idx < 8) ? idx : idx + 8) + ((lane < 16) ? 0 : 8);  // B: K rows 0-7,16-23 / 8-15,24-31
      bband[idx] = (Kb >= m && Kb <= m + 4) ? (_Float16)1.0f : (_Float16)0.0f;
    }
  }

  v8f stdp, stdt;
#pragma unroll
  for (int field = 0; field < 2; ++field) {
    // Zero X1+X2 (2048 f16 = 512 u64) so halo padding & unused rows/cols are 0.
    unsigned long long* z = (unsigned long long*)X1;
#pragma unroll
    for (int i = lane; i < 512; i += 32) z[i] = 0ull;

    // Fill the 20x20 halo (rows y0-2..y0+17, cols x0-2..x0+17), zero-padded.
    for (int p = lane; p < 400; p += 32) {
      const int r = p / 20, c = p - r * 20;
      const int y = y0 - 2 + r, x = x0 - 2 + c;
      if ((unsigned)y < DIM_H && (unsigned)x < DIM_W) {
        float v;
        if (field == 0) {
          const float* pp = pred + (size_t)b * DIM_C * CH_STRIDE
                                 + (size_t)y * DIM_W + x;
          float best = pp[0];
          int bi = 0;
#pragma unroll
          for (int ch = 1; ch < DIM_C; ++ch) {
            const float vv = pp[(size_t)ch * CH_STRIDE];
            if (vv > best) { best = vv; bi = ch; }   // strict '>' keeps first max
          }
          v = (float)bi;
        } else {
          v = (float)tgt[((size_t)b * DIM_H + y) * DIM_W + x];
        }
        const _Float16 hv = (_Float16)v;
        X1[c * 32 + r] = hv;          // column-major: col*32 + row
        X2[c * 32 + r] = hv * hv;
      }
    }
    asm volatile("s_wait_dscnt 0" ::: "memory");

    const v8f S1 = box_filter_16x16(X1, Ys, aband, bband, lane);
    const v8f S2 = box_filter_16x16(X2, Ys, aband, bband, lane);

    v8f sd;
#pragma unroll
    for (int r = 0; r < 8; ++r) {
      const float s1 = S1[r], s2 = S2[r];
      const float var = (s2 - s1 * s1 / 25.0f) / 24.0f;  // unbiased, n=25
      sd[r] = sqrtf(fmaxf(var, 0.0f));
    }
    if (field == 0) stdp = sd; else stdt = sd;
  }

  float part = 0.0f;
#pragma unroll
  for (int r = 0; r < 8; ++r) {
    const float d = stdp[r] - stdt[r];
    part += d * d;
  }
#pragma unroll
  for (int off = 16; off > 0; off >>= 1)
    part += __shfl_xor(part, off, 32);
  if (lane == 0)
    atomicAdd(out, part * (1.0f / 8388608.0f));  // mean over 8*1024*1024
}

__global__ void dcl_zero(float* o) {
  if (threadIdx.x == 0) o[0] = 0.0f;
}

extern "C" void kernel_launch(void* const* d_in, const int* in_sizes, int n_in,
                              void* d_out, int out_size, void* d_ws, size_t ws_size,
                              hipStream_t stream) {
  const float*     pred = (const float*)d_in[0];       // [8,6,1024,1024] f32
  const long long* tgt  = (const long long*)d_in[1];   // [8,1024,1024] i64
  float*           out  = (float*)d_out;               // scalar f32
  (void)in_sizes; (void)n_in; (void)out_size; (void)d_ws; (void)ws_size;

  dcl_zero<<<1, 1, 0, stream>>>(out);
  // 32768 tiles (8 batches * 64 * 64), 8 waves/block -> 4096 blocks
  dcl_kernel<<<4096, 256, 0, stream>>>(pred, tgt, out);
}
